// Speller_27376121545374
// MI455X (gfx1250) — compile-verified, matrix operand order checked by reading, not run on hardware
//
#include <hip/hip_runtime.h>
#include <hip/hip_bf16.h>
#include <math.h>

typedef __bf16 bf16;
typedef __attribute__((ext_vector_type(16))) __bf16 v16bf;
typedef __attribute__((ext_vector_type(8)))  float  v8f;

union FragBF { v16bf v; float4 f4[2]; };

#define B_   128
#define S_   600
#define T_   200
#define ENC_ 512
#define P_   128
#define E_   256
#define H_   512
#define V_   31

// ---------------- f32 -> bf16 convert ----------------
__global__ void k_f32_to_bf16(const float* __restrict__ src, bf16* __restrict__ dst, int n) {
    int i = blockIdx.x * blockDim.x + threadIdx.x;
    if (i < n) dst[i] = (bf16)src[i];
}

// ---------------- pack weight (N x K row-major, optional K-split [Wa|Wb]) into
// WMMA B-fragment tiles: [ktile][ntile][lane 0..31][elem 0..15] bf16 ----------
__global__ void k_pack(const float* __restrict__ Wa, int Ka,
                       const float* __restrict__ Wb, int Kb,
                       int N, int Ntiles, bf16* __restrict__ dst, int total) {
    int idx = blockIdx.x * blockDim.x + threadIdx.x;
    if (idx >= total) return;
    int e     = idx & 15;
    int lane  = (idx >> 4) & 31;
    int tile  = idx >> 9;
    int ntile = tile % Ntiles;
    int ktile = tile / Ntiles;
    int n = ntile * 16 + (lane & 15);
    int k = ktile * 32 + ((lane >> 4) * 8) + (e & 7) + ((e & 8) ? 16 : 0);
    float v = 0.f;
    int K = Ka + Kb;
    if (n < N && k < K) v = (k < Ka) ? Wa[(size_t)n * Ka + k] : Wb[(size_t)n * Kb + (k - Ka)];
    dst[idx] = (bf16)v;
}

// ---------------- generic bf16 WMMA GEMM with register N-blocking and
// K-iteration double buffering (loads for kt+1 overlap the NB WMMAs of kt) -----
// C(MxN,f32) = A(MxK,bf16,row-major lda) x packed-B + bias.
// ntiles must be a multiple of NB; M must be a multiple of 16.
template <int NB>
__global__ void k_wmma_gemm(const bf16* __restrict__ A, int lda,
                            const bf16* __restrict__ Bp,
                            const float* __restrict__ bias,
                            float* __restrict__ C, int ldc,
                            int N, int ktiles, int ntiles) {
    int ngroups = ntiles / NB;
    int tile    = blockIdx.x;
    int group   = tile % ngroups;
    int mtile   = tile / ngroups;
    int ntile0  = group * NB;
    int lane    = threadIdx.x;        // 0..31 (wave32)
    int nloc    = lane & 15;
    int half    = lane >> 4;

    v8f acc[NB];
#pragma unroll
    for (int nb = 0; nb < NB; ++nb) {
        int n = (ntile0 + nb) * 16 + nloc;
        float bv = (bias != nullptr && n < N) ? bias[n] : 0.f;
#pragma unroll
        for (int r = 0; r < 8; ++r) acc[nb][r] = bv;
    }

    // A: lane (0..15) holds row M=lane K[0..7,16..23]; lane (16..31) same row, K[8..15,24..31]
    const bf16* arow  = A + (size_t)(mtile * 16 + nloc) * lda + half * 8;
    const bf16* bbase = Bp + (((size_t)ntile0 * 32) + lane) * 16;
    size_t bstride = (size_t)ntiles * 32 * 16;   // elements per ktile of packed B

    FragBF fa, fb[NB];
    {
        fa.f4[0] = *(const float4*)(arow);
        fa.f4[1] = *(const float4*)(arow + 16);
#pragma unroll
        for (int nb = 0; nb < NB; ++nb) {
            const bf16* bp = bbase + (size_t)nb * 32 * 16;
            fb[nb].f4[0] = *(const float4*)(bp);
            fb[nb].f4[1] = *(const float4*)(bp + 8);
        }
    }

    for (int kt = 0; kt < ktiles; ++kt) {
        // issue next iteration's loads before the math so they overlap the WMMAs
        int ktn = (kt + 1 < ktiles) ? kt + 1 : kt;
        FragBF fa_n, fb_n[NB];
        const bf16* apn = arow + (size_t)ktn * 32;
        fa_n.f4[0] = *(const float4*)(apn);
        fa_n.f4[1] = *(const float4*)(apn + 16);
        const bf16* bbn = bbase + (size_t)ktn * bstride;
#pragma unroll
        for (int nb = 0; nb < NB; ++nb) {
            const bf16* bp = bbn + (size_t)nb * 32 * 16;
            fb_n[nb].f4[0] = *(const float4*)(bp);
            fb_n[nb].f4[1] = *(const float4*)(bp + 8);
        }

#pragma unroll
        for (int nb = 0; nb < NB; ++nb) {
            acc[nb] = __builtin_amdgcn_wmma_f32_16x16x32_bf16(
                /*neg_a=*/false, fa.v, /*neg_b=*/false, fb[nb].v,
                /*c_mod=*/(short)0, acc[nb], /*reuse_a=*/false, /*reuse_b=*/false);
        }

        fa = fa_n;
#pragma unroll
        for (int nb = 0; nb < NB; ++nb) fb[nb] = fb_n[nb];
    }

#pragma unroll
    for (int nb = 0; nb < NB; ++nb) {
        int n = (ntile0 + nb) * 16 + nloc;
        if (n < N) {
            float* crow = C + (size_t)(mtile * 16 + half * 8) * ldc + n;
#pragma unroll
            for (int r = 0; r < 8; ++r) crow[(size_t)r * ldc] = acc[nb][r];
        }
    }
}

// ---------------- LSTM pointwise: gates(B x 4H, order i,f,g,o) -> c,h ----------
__global__ void k_lstm_pw(const float* __restrict__ g, float* __restrict__ cst,
                          bf16* __restrict__ dA, int strideA,
                          bf16* __restrict__ dB, int strideB) {
    int idx = blockIdx.x * blockDim.x + threadIdx.x;
    if (idx >= B_ * H_) return;
    int b = idx / H_, j = idx % H_;
    const float* gr = g + (size_t)b * 4 * H_;
    float i  = gr[j];
    float f  = gr[H_ + j];
    float gg = gr[2 * H_ + j];
    float o  = gr[3 * H_ + j];
    float si = 1.f / (1.f + __expf(-i));
    float sf = 1.f / (1.f + __expf(-f));
    float so = 1.f / (1.f + __expf(-o));
    float c  = sf * cst[idx] + si * tanhf(gg);
    cst[idx] = c;
    float h  = so * tanhf(c);
    dA[(size_t)b * strideA + j] = (bf16)h;
    dB[(size_t)b * strideB + j] = (bf16)h;
}

// ---------------- build cell0 input [emb[tok] | ctx] (h part persists) --------
__global__ void k_build_x0(const int* __restrict__ y, const float* __restrict__ emb,
                           const float* __restrict__ ctx, bf16* __restrict__ x0, int t) {
    int idx = blockIdx.x * blockDim.x + threadIdx.x;
    if (idx >= B_ * (E_ + P_)) return;
    int b = idx / (E_ + P_), j = idx % (E_ + P_);
    float v;
    if (j < E_) {
        int tok = (t == 0) ? 1 : y[(size_t)b * T_ + (t - 1)];
        v = emb[(size_t)tok * E_ + j];
    } else {
        v = ctx[(size_t)b * P_ + (j - E_)];
    }
    x0[(size_t)b * (E_ + P_ + H_) + j] = (bf16)v;
}

// ---------------- initial h/c state scatter ----------------
__global__ void k_init_states(const float* __restrict__ h0, const float* __restrict__ c0,
                              float* __restrict__ cst,
                              bf16* __restrict__ x0, bf16* __restrict__ x1, bf16* __restrict__ x2) {
    int idx = blockIdx.x * blockDim.x + threadIdx.x;
    if (idx >= 3 * B_ * H_) return;
    int cell = idx / (B_ * H_);
    int rem  = idx % (B_ * H_);
    int b = rem / H_, j = rem % H_;
    cst[idx] = c0[idx];
    bf16 h = (bf16)h0[idx];
    if (cell == 0)      x0[(size_t)b * 896  + 384 + j] = h;
    else if (cell == 1) x1[(size_t)b * 1024 + 512 + j] = h;
    else                x2[(size_t)b * 1024 + 512 + j] = h;
}

__global__ void k_zero_f32(float* __restrict__ p, int n) {
    int i = blockIdx.x * blockDim.x + threadIdx.x;
    if (i < n) p[i] = 0.f;
}

__global__ void k_bias_sum(const float* __restrict__ a, const float* __restrict__ b,
                           float* __restrict__ o, int n) {
    int i = blockIdx.x * blockDim.x + threadIdx.x;
    if (i < n) o[i] = a[i] + b[i];
}

// ---------------- exact GELU -> bf16 ----------------
__global__ void k_gelu(const float* __restrict__ in, bf16* __restrict__ out, int n) {
    int i = blockIdx.x * blockDim.x + threadIdx.x;
    if (i >= n) return;
    float x = in[i];
    out[i] = (bf16)(0.5f * x * (1.f + erff(x * 0.70710678118654752f)));
}

// ---------------- attention: logits, masked softmax, context; one block per b -
__global__ void k_attention(const float* __restrict__ key, const float* __restrict__ value,
                            const float* __restrict__ q, const int* __restrict__ enc_lens,
                            float* __restrict__ attn_out, float* __restrict__ ctx,
                            bf16* __restrict__ cdn_in, int t) {
    __shared__ float qs[P_];
    __shared__ float lg[S_];
    __shared__ float red[256];
    __shared__ float smax, ssum;
    int b = blockIdx.x;
    int tid = threadIdx.x;
    if (tid < P_) qs[tid] = q[(size_t)b * P_ + tid];
    __syncthreads();
    int len = enc_lens[b];
    const float* kb = key + (size_t)b * S_ * P_;
    float lmax = -3.4e38f;
    for (int s = tid; s < S_; s += 256) {
        float d = 0.f;
        const float* kr = kb + (size_t)s * P_;
        for (int p = 0; p < P_; ++p) d += kr[p] * qs[p];
        d *= 0.08838834764831845f;   // 1/sqrt(128)
        if (s >= len) d = -3.4e38f;
        lg[s] = d;
        lmax = fmaxf(lmax, d);
    }
    red[tid] = lmax;
    __syncthreads();
    if (tid == 0) {
        float m = -3.4e38f;
        for (int i = 0; i < 256; ++i) m = fmaxf(m, red[i]);
        smax = m;
    }
    __syncthreads();
    float lsum = 0.f;
    for (int s = tid; s < S_; s += 256) {
        float e = (lg[s] <= -3.0e38f) ? 0.f : __expf(lg[s] - smax);
        lg[s] = e;
        lsum += e;
    }
    red[tid] = lsum;
    __syncthreads();
    if (tid == 0) {
        float acc = 0.f;
        for (int i = 0; i < 256; ++i) acc += red[i];
        ssum = acc;
    }
    __syncthreads();
    float inv = 1.f / ssum;
    float* ao = attn_out + (size_t)b * T_ * S_ + (size_t)t * S_;
    for (int s = tid; s < S_; s += 256) {
        float w = lg[s] * inv;
        lg[s] = w;
        ao[s] = w;
    }
    __syncthreads();
    const float* vb = value + (size_t)b * S_ * P_;
    for (int p = tid; p < P_; p += 256) {
        float a = 0.f;
        for (int s = 0; s < S_; ++s) a += lg[s] * vb[(size_t)s * P_ + p];
        ctx[(size_t)b * P_ + p] = a;
        cdn_in[(size_t)b * 640 + 512 + p] = (bf16)a;
    }
}

extern "C" void kernel_launch(void* const* d_in, const int* in_sizes, int n_in,
                              void* d_out, int out_size, void* d_ws, size_t ws_size,
                              hipStream_t stream) {
    const float* x        = (const float*)d_in[0];
    const int*   y        = (const int*)d_in[1];
    const int*   enc_lens = (const int*)d_in[2];
    const float* h0       = (const float*)d_in[3];
    const float* c0       = (const float*)d_in[4];
    const float* emb      = (const float*)d_in[5];
    const float* W_ih0 = (const float*)d_in[6];  const float* W_hh0 = (const float*)d_in[7];
    const float* b_ih0 = (const float*)d_in[8];  const float* b_hh0 = (const float*)d_in[9];
    const float* W_ih1 = (const float*)d_in[10]; const float* W_hh1 = (const float*)d_in[11];
    const float* b_ih1 = (const float*)d_in[12]; const float* b_hh1 = (const float*)d_in[13];
    const float* W_ih2 = (const float*)d_in[14]; const float* W_hh2 = (const float*)d_in[15];
    const float* b_ih2 = (const float*)d_in[16]; const float* b_hh2 = (const float*)d_in[17];
    const float* KW_w  = (const float*)d_in[18]; const float* KW_b  = (const float*)d_in[19];
    const float* VW_w  = (const float*)d_in[20]; const float* VW_b  = (const float*)d_in[21];
    const float* QW_w  = (const float*)d_in[22]; const float* QW_b  = (const float*)d_in[23];
    const float* out_w = (const float*)d_in[24]; const float* out_b = (const float*)d_in[25];
    const float* char_w = (const float*)d_in[26]; const float* char_b = (const float*)d_in[27];

    float* out_pred = (float*)d_out;                              // (B,T,V)
    float* out_attn = (float*)d_out + (size_t)B_ * T_ * V_;       // (B,T,S)

    // ---- workspace carve-out ----
    size_t off = 0;
    auto alloc = [&](size_t bytes) -> char* {
        char* p = (char*)d_ws + off;
        off += (bytes + 255) & ~(size_t)255;
        return p;
    };
    bf16*  x_bf   = (bf16*)alloc((size_t)B_ * S_ * ENC_ * 2);
    float* key    = (float*)alloc((size_t)B_ * S_ * P_ * 4);
    float* value  = (float*)alloc((size_t)B_ * S_ * P_ * 4);
    bf16*  pG0    = (bf16*)alloc((size_t)28 * 128 * 512 * 2);
    bf16*  pG1    = (bf16*)alloc((size_t)32 * 128 * 512 * 2);
    bf16*  pG2    = (bf16*)alloc((size_t)32 * 128 * 512 * 2);
    bf16*  pQW    = (bf16*)alloc((size_t)16 * 8 * 512 * 2);
    bf16*  pOUT   = (bf16*)alloc((size_t)20 * 16 * 512 * 2);
    bf16*  pCHAR  = (bf16*)alloc((size_t)8 * 2 * 512 * 2);
    bf16*  pKW    = (bf16*)alloc((size_t)16 * 8 * 512 * 2);
    bf16*  pVW    = (bf16*)alloc((size_t)16 * 8 * 512 * 2);
    float* bsum0  = (float*)alloc(2048 * 4);
    float* bsum1  = (float*)alloc(2048 * 4);
    float* bsum2  = (float*)alloc(2048 * 4);
    bf16*  x0     = (bf16*)alloc((size_t)B_ * 896 * 2);
    bf16*  x1     = (bf16*)alloc((size_t)B_ * 1024 * 2);
    bf16*  x2     = (bf16*)alloc((size_t)B_ * 1024 * 2);
    bf16*  cdn_in = (bf16*)alloc((size_t)B_ * 640 * 2);
    float* hid_pre = (float*)alloc((size_t)B_ * 256 * 4);
    bf16*  hid    = (bf16*)alloc((size_t)B_ * 256 * 2);
    float* gates  = (float*)alloc((size_t)B_ * 2048 * 4);
    float* cst    = (float*)alloc((size_t)3 * B_ * H_ * 4);
    float* ctxbuf = (float*)alloc((size_t)B_ * P_ * 4);
    float* qbuf   = (float*)alloc((size_t)B_ * P_ * 4);

    auto gemm = [&](const bf16* A, int lda, const bf16* Bp, const float* bias,
                    float* C, int ldc, int M, int N, int K) {
        int mt = M / 16, nt = (N + 15) / 16;
        if (nt % 4 == 0) {
            k_wmma_gemm<4><<<dim3(mt * (nt / 4)), dim3(32), 0, stream>>>(
                A, lda, Bp, bias, C, ldc, N, K / 32, nt);
        } else if (nt % 2 == 0) {
            k_wmma_gemm<2><<<dim3(mt * (nt / 2)), dim3(32), 0, stream>>>(
                A, lda, Bp, bias, C, ldc, N, K / 32, nt);
        } else {
            k_wmma_gemm<1><<<dim3(mt * nt), dim3(32), 0, stream>>>(
                A, lda, Bp, bias, C, ldc, N, K / 32, nt);
        }
    };
    auto cdiv = [](int a, int b) { return (a + b - 1) / b; };

    // ---- one-time prep (re-done every call: deterministic) ----
    {
        int n = B_ * S_ * ENC_;
        k_f32_to_bf16<<<cdiv(n, 256), 256, 0, stream>>>(x, x_bf, n);
    }
    k_pack<<<cdiv(65536, 256), 256, 0, stream>>>(KW_w, 512, nullptr, 0, 128, 8, pKW, 65536);
    k_pack<<<cdiv(65536, 256), 256, 0, stream>>>(VW_w, 512, nullptr, 0, 128, 8, pVW, 65536);
    k_pack<<<cdiv(1835008, 256), 256, 0, stream>>>(W_ih0, 384, W_hh0, 512, 2048, 128, pG0, 1835008);
    k_pack<<<cdiv(2097152, 256), 256, 0, stream>>>(W_ih1, 512, W_hh1, 512, 2048, 128, pG1, 2097152);
    k_pack<<<cdiv(2097152, 256), 256, 0, stream>>>(W_ih2, 512, W_hh2, 512, 2048, 128, pG2, 2097152);
    k_pack<<<cdiv(65536, 256), 256, 0, stream>>>(QW_w, 512, nullptr, 0, 128, 8, pQW, 65536);
    k_pack<<<cdiv(163840, 256), 256, 0, stream>>>(out_w, 640, nullptr, 0, 256, 16, pOUT, 163840);
    k_pack<<<cdiv(8192, 256), 256, 0, stream>>>(char_w, 256, nullptr, 0, 31, 2, pCHAR, 8192);
    k_bias_sum<<<cdiv(2048, 256), 256, 0, stream>>>(b_ih0, b_hh0, bsum0, 2048);
    k_bias_sum<<<cdiv(2048, 256), 256, 0, stream>>>(b_ih1, b_hh1, bsum1, 2048);
    k_bias_sum<<<cdiv(2048, 256), 256, 0, stream>>>(b_ih2, b_hh2, bsum2, 2048);
    k_init_states<<<cdiv(3 * B_ * H_, 256), 256, 0, stream>>>(h0, c0, cst, x0, x1, x2);
    k_zero_f32<<<cdiv(B_ * P_, 256), 256, 0, stream>>>(ctxbuf, B_ * P_);

    // key/value projections: (B*S x 512) @ packed(512 x 128) via WMMA
    gemm(x_bf, ENC_, pKW, KW_b, key, P_, B_ * S_, P_, ENC_);
    gemm(x_bf, ENC_, pVW, VW_b, value, P_, B_ * S_, P_, ENC_);

    // ---- sequential decode: 200 steps ----
    for (int t = 0; t < T_; ++t) {
        k_build_x0<<<cdiv(B_ * 384, 256), 256, 0, stream>>>(y, emb, ctxbuf, x0, t);

        // cell 0: x0 (B x 896) -> gates; h0 -> x1[:, :512] and x0[:, 384:896]
        gemm(x0, 896, pG0, bsum0, gates, 2048, B_, 2048, 896);
        k_lstm_pw<<<cdiv(B_ * H_, 256), 256, 0, stream>>>(gates, cst, x1, 1024, x0 + 384, 896);

        // cell 1: x1 (B x 1024) -> gates; h1 -> x2[:, :512] and x1[:, 512:]
        gemm(x1, 1024, pG1, bsum1, gates, 2048, B_, 2048, 1024);
        k_lstm_pw<<<cdiv(B_ * H_, 256), 256, 0, stream>>>(gates, cst + B_ * H_, x2, 1024, x1 + 512, 1024);

        // cell 2: x2 (B x 1024) -> gates; h2 -> cdn_in[:, :512] and x2[:, 512:]
        gemm(x2, 1024, pG2, bsum2, gates, 2048, B_, 2048, 1024);
        k_lstm_pw<<<cdiv(B_ * H_, 256), 256, 0, stream>>>(gates, cst + 2 * B_ * H_, cdn_in, 640, x2 + 512, 1024);

        // query: h2 (first 512 cols of cdn_in) @ QW^T
        gemm(cdn_in, 640, pQW, QW_b, qbuf, P_, B_, P_, H_);

        // attention: softmax over S, context -> ctxbuf + cdn_in[:, 512:640]; weights -> d_out
        k_attention<<<dim3(B_), dim3(256), 0, stream>>>(key, value, qbuf, enc_lens,
                                                        out_attn, ctxbuf, cdn_in, t);

        // CDN head: [h2|ctx] (B x 640) @ out_w^T -> GELU -> char proj into d_out
        gemm(cdn_in, 640, pOUT, out_b, hid_pre, 256, B_, 256, 640);
        k_gelu<<<cdiv(B_ * 256, 256), 256, 0, stream>>>(hid_pre, hid, B_ * 256);
        gemm(hid, 256, pCHAR, char_b, out_pred + (size_t)t * V_, T_ * V_, B_, V_, 256);
    }
    (void)in_sizes; (void)n_in; (void)out_size; (void)ws_size;
}
